// BiLSTM_CRF_85031762526518
// MI455X (gfx1250) — compile-verified, matrix operand order checked by reading, not run on hardware
//
#include <hip/hip_runtime.h>
#include <hip/hip_bf16.h>

// ---------------- CDNA5 WMMA types ----------------
typedef __attribute__((ext_vector_type(16))) __bf16 bf16x16;
typedef __attribute__((ext_vector_type(8)))  float  floatx8;

union ABFrag {            // 16 bf16 = 32B = 8 VGPRs (A or B operand)
    bf16x16 v;
    unsigned short u[16];
    uint4 q[2];
};
union CFrag {             // 8 f32 = 8 VGPRs (C/D operand)
    floatx8 v;
    float f[8];
    float4 q[2];
};

__device__ __forceinline__ unsigned short f32_to_bf16(float f) {
    unsigned int u = __float_as_uint(f);
    u += 0x7FFFu + ((u >> 16) & 1u);     // round-to-nearest-even
    return (unsigned short)(u >> 16);
}
__device__ __forceinline__ float sigm(float x) { return 1.0f / (1.0f + __expf(-x)); }

// Problem constants
#define BB 32
#define TT 256
#define EE 256
#define HH 256
#define NT 12
#define G4H 1024           // 4*H

// ---------------------------------------------------------------
// K0: W_ih fp32->bf16 (row-major), W_hh fp32->bf16 repacked into
// WMMA-B fragment order [dir][jg][gate][kt][lane][16], and
// embedding gather (bf16, row = t*32+b).
// ---------------------------------------------------------------
__global__ void k0_prep(const int* __restrict__ sent,
                        const float* __restrict__ emb,
                        const float* __restrict__ Wih_f, const float* __restrict__ Whh_f,
                        const float* __restrict__ Wih_b, const float* __restrict__ Whh_b,
                        unsigned short* __restrict__ WihX,   // [2][1024*256] row-major
                        unsigned short* __restrict__ WhhP,   // [2][16][4][8][32][16] frag order
                        unsigned short* __restrict__ Xb)     // [8192][256]
{
    const int W1 = G4H * HH;          // 262144
    int id = blockIdx.x * blockDim.x + threadIdx.x;
    if (id < W1) {
        WihX[id] = f32_to_bf16(Wih_f[id]);
    } else if (id < 2 * W1) {
        WihX[id] = f32_to_bf16(Wih_b[id - W1]);
    } else if (id < 4 * W1) {
        // fragment-order repack of W_hh (B operand of h @ W_hh^T)
        int o = id - 2 * W1;                  // 0 .. 524287
        int e    = o & 15;
        int lane = (o >> 4) & 31;
        int kt   = (o >> 9) & 7;
        int gate = (o >> 12) & 3;
        int jg   = (o >> 14) & 15;
        int dir  = o >> 18;
        int n  = gate * HH + jg * 16 + (lane & 15);               // gate column
        int k  = kt * 32 + 8 * (lane >> 4) + (e & 7) + ((e >= 8) ? 16 : 0);
        const float* src = dir ? Whh_b : Whh_f;
        WhhP[o] = f32_to_bf16(src[n * HH + k]);
    } else {
        int e = id - 4 * W1;                 // 0 .. 8192*256-1
        if (e < BB * TT * EE) {
            int r = e >> 8;                  // row = t*32+b
            int k = e & 255;
            int b = r & 31, t = r >> 5;
            int tok = sent[b * TT + t];
            Xb[e] = f32_to_bf16(emb[tok * EE + k]);
        }
    }
}

// ---------------------------------------------------------------
// K1: XG[dir][t][n][b] = (X @ W_ih^T)[t*32+b][n] + b_ih[n] + b_hh[n]
// M=8192 (512 tiles), N=1024 (64 tiles), K=256 (8 wmma) per direction.
// One 16x16 tile per wave; bf16 WMMA, fp32 accumulate.
// ---------------------------------------------------------------
__global__ void k1_input_gemm(const unsigned short* __restrict__ Xb,
                              const unsigned short* __restrict__ WihX,
                              const float* __restrict__ bih_f, const float* __restrict__ bhh_f,
                              const float* __restrict__ bih_b, const float* __restrict__ bhh_b,
                              float* __restrict__ XGT)
{
    int wid  = blockIdx.x * (blockDim.x >> 5) + (threadIdx.x >> 5);
    int lane = threadIdx.x & 31;
    int mt   = wid & 511;
    int rest = wid >> 9;
    int nt   = rest & 63;
    int dir  = rest >> 6;

    int ln = lane & 15;          // A-row / B-col within tile
    int lh = lane >> 4;          // K-half selector

    int n     = nt * 16 + ln;                  // gate column
    int m_row = mt * 16 + ln;                  // X row (= t*32 + b)
    const unsigned short* W = WihX + dir * (G4H * HH);

    float bias = dir ? (bih_b[n] + bhh_b[n]) : (bih_f[n] + bhh_f[n]);

    CFrag acc;
#pragma unroll
    for (int v = 0; v < 8; ++v) acc.f[v] = 0.0f;

#pragma unroll
    for (int kt = 0; kt < 8; ++kt) {
        int kb = kt * 32 + 8 * lh;
        ABFrag a, b;
        const unsigned short* ap = Xb + m_row * EE + kb;
        a.q[0] = *(const uint4*)(ap);
        a.q[1] = *(const uint4*)(ap + 16);
        const unsigned short* bp = W + n * HH + kb;
        b.q[0] = *(const uint4*)(bp);
        b.q[1] = *(const uint4*)(bp + 16);
        acc.v = __builtin_amdgcn_wmma_f32_16x16x32_bf16(
            false, a.v, false, b.v, (short)0, acc.v, false, false);
    }
#pragma unroll
    for (int v = 0; v < 8; ++v) acc.f[v] += bias;

    // D layout: lane n fixed, M = 8*lh + v -> b consecutive.  Tile rows share one t.
    int r0 = mt * 16;
    int t  = r0 >> 5;
    int bbase = (r0 & 31) + 8 * lh;
    float* dp = XGT + (((dir * TT + t) * G4H + n) * BB + bbase);
    *(float4*)dp       = acc.q[0];
    *(float4*)(dp + 4) = acc.q[1];
}

// ---------------------------------------------------------------
// K2: sequential LSTM recurrence.  One block per direction, 32 waves.
// Wave w: mh = w&1 (batch half), jg = w>>1 (16 hidden units).
// c-tile lives in registers for all 256 steps; h ping-pongs via LDS bf16.
// W_hh B-fragments stream fully-coalesced from the fragment-order pack.
// ---------------------------------------------------------------
__global__ void __launch_bounds__(1024)
k2_lstm_rec(const unsigned short* __restrict__ WhhP,
            const float* __restrict__ XGT,
            const float* __restrict__ h0, const float* __restrict__ c0,
            float* __restrict__ HS)                 // [2][T][H][B]
{
    __shared__ unsigned short hbuf[2][BB * HH];     // 32 KB ping-pong

    const int dir  = blockIdx.x;
    const int tid  = threadIdx.x;
    const int wid  = tid >> 5;
    const int lane = tid & 31;
    const int mh = wid & 1;
    const int jg = wid >> 1;
    const int j0 = jg * 16;
    const int ln = lane & 15;
    const int lh = lane >> 4;

    for (int i = tid; i < BB * HH; i += 1024)
        hbuf[0][i] = f32_to_bf16(h0[dir * BB * HH + i]);

    // cell state tile in registers (C/D layout: lane j fixed, m = 8*lh+v)
    CFrag c;
#pragma unroll
    for (int v = 0; v < 8; ++v)
        c.f[v] = c0[dir * BB * HH + (mh * 16 + 8 * lh + v) * HH + j0 + ln];

    __syncthreads();

    // this wave's contiguous 32KB fragment block: [gate][kt][lane][16]
    const unsigned short* Wp = WhhP + (size_t)(dir * 16 + jg) * (4 * 8 * 32 * 16) + lane * 16;
    const float*          XG = XGT + dir * (TT * G4H * BB);
    int cur = 0;

    for (int s = 0; s < TT; ++s) {
        const int t = dir ? (TT - 1 - s) : s;

        // C fragments = precomputed input gates (i,f,g,o)
        CFrag g[4];
#pragma unroll
        for (int gg = 0; gg < 4; ++gg) {
            const float* cp = XG + ((t * G4H) + gg * HH + j0 + ln) * BB + mh * 16 + 8 * lh;
            g[gg].q[0] = *(const float4*)cp;
            g[gg].q[1] = *(const float4*)(cp + 4);
        }

        // prefetch next step's gate pre-activations while WMMAs run
        if (s + 1 < TT) {
            const int tn = dir ? (t - 1) : (t + 1);
#pragma unroll
            for (int gg = 0; gg < 4; ++gg)
                __builtin_prefetch(XG + ((tn * G4H) + gg * HH + j0 + ln) * BB + mh * 16 + 8 * lh, 0, 3);
        }

        // h_{t-1} @ W_hh^T : K=256, 8 k-tiles, A from LDS, B streamed from L2
#pragma unroll
        for (int kt = 0; kt < 8; ++kt) {
            ABFrag a;
            const unsigned short* ap = &hbuf[cur][(mh * 16 + ln) * HH + kt * 32 + 8 * lh];
            a.q[0] = *(const uint4*)(ap);
            a.q[1] = *(const uint4*)(ap + 16);
#pragma unroll
            for (int gg = 0; gg < 4; ++gg) {
                ABFrag b;
                const unsigned short* bp = Wp + (gg * 8 + kt) * (32 * 16);
                b.q[0] = *(const uint4*)(bp);
                b.q[1] = *(const uint4*)(bp + 8);
                g[gg].v = __builtin_amdgcn_wmma_f32_16x16x32_bf16(
                    false, a.v, false, b.v, (short)0, g[gg].v, false, false);
            }
        }

        // gate nonlinearity fully in registers (tiles share lane layout)
        CFrag h;
#pragma unroll
        for (int v = 0; v < 8; ++v) {
            float iv = sigm(g[0].f[v]);
            float fv = sigm(g[1].f[v]);
            float gv = tanhf(g[2].f[v]);
            float ov = sigm(g[3].f[v]);
            c.f[v] = fv * c.f[v] + iv * gv;
            h.f[v] = ov * tanhf(c.f[v]);
        }

        // publish h: bf16 to next LDS buffer + fp32 to global HS[dir][t][j][b]
        const int nxt = cur ^ 1;
#pragma unroll
        for (int v = 0; v < 8; ++v) {
            int brow = mh * 16 + 8 * lh + v;
            hbuf[nxt][brow * HH + j0 + ln] = f32_to_bf16(h.f[v]);
        }
        float* hp = HS + (((dir * TT + t) * HH + j0 + ln) * BB + mh * 16 + 8 * lh);
        *(float4*)hp       = h.q[0];
        *(float4*)(hp + 4) = h.q[1];

        __syncthreads();
        cur = nxt;
    }
}

// ---------------------------------------------------------------
// K3: feats[b][t][n] = concat(h_f,h_b)[b][t] . W_out[n] + b_out[n]
// grid = T*NT blocks of 32 (lane = b) -> coalesced HS reads
// ---------------------------------------------------------------
__global__ void k3_feats(const float* __restrict__ HS,
                         const float* __restrict__ W_out,
                         const float* __restrict__ b_out,
                         float* __restrict__ feats)
{
    int t = blockIdx.x / NT;
    int n = blockIdx.x % NT;
    int b = threadIdx.x;

    const float* wf = W_out + n * (2 * HH);
    const float* hf = HS + (t * HH) * BB + b;                     // forward dir
    const float* hb = HS + (TT * HH * BB) + (t * HH) * BB + b;    // backward dir
    float acc = b_out[n];
    for (int j = 0; j < HH; ++j) acc += wf[j]       * hf[j * BB];
    for (int j = 0; j < HH; ++j) acc += wf[HH + j]  * hb[j * BB];
    feats[(b * TT + t) * NT + n] = acc;
}

// ---------------------------------------------------------------
// K4: per-batch CRF forward (logZ) + gold score.  One wave per b.
// ---------------------------------------------------------------
__global__ void k4_crf(const float* __restrict__ feats,
                       const float* __restrict__ trans,
                       const int* __restrict__ tags,
                       float* __restrict__ partial)
{
    __shared__ float tr[NT * NT];
    __shared__ float fvs[NT];

    const int b = blockIdx.x;
    const int lane = threadIdx.x;
    for (int i = lane; i < NT * NT; i += 32) tr[i] = trans[i];

    float fv = (lane == 0) ? 0.0f : -10000.0f;   // fv[START]=0
    __syncthreads();

    const float* fb = feats + b * TT * NT;
    for (int t = 0; t < TT; ++t) {
        if (lane < NT) fvs[lane] = fv;
        __syncthreads();
        float nfv = fv;
        if (lane < NT) {
            float m = -3.0e38f;
            for (int i = 0; i < NT; ++i) m = fmaxf(m, fvs[i] + tr[lane * NT + i]);
            float s = 0.0f;
            for (int i = 0; i < NT; ++i) s += __expf(fvs[i] + tr[lane * NT + i] - m);
            nfv = fb[t * NT + lane] + m + __logf(s);
        }
        __syncthreads();
        fv = nfv;
    }
    if (lane < NT) fvs[lane] = fv + tr[1 * NT + lane];   // + trans[STOP][i]
    __syncthreads();

    if (lane == 0) {
        float m = -3.0e38f;
        for (int i = 0; i < NT; ++i) m = fmaxf(m, fvs[i]);
        float s = 0.0f;
        for (int i = 0; i < NT; ++i) s += __expf(fvs[i] - m);
        float logz = m + __logf(s);

        const int* tg = tags + b * TT;
        int prev = 0;                                    // START
        float gold = 0.0f;
        for (int t = 0; t < TT; ++t) {
            int tag = tg[t];
            gold += fb[t * NT + tag] + tr[tag * NT + prev];
            prev = tag;
        }
        gold += tr[1 * NT + prev];                       // trans[STOP][last]
        partial[b] = logz - gold;
    }
}

// K5: sum 32 partials -> scalar
__global__ void k5_reduce(const float* __restrict__ partial, float* __restrict__ out)
{
    float v = partial[threadIdx.x & 31];
    for (int off = 16; off > 0; off >>= 1) v += __shfl_down(v, off, 32);
    if (threadIdx.x == 0) out[0] = v;
}

// ---------------------------------------------------------------
extern "C" void kernel_launch(void* const* d_in, const int* in_sizes, int n_in,
                              void* d_out, int out_size, void* d_ws, size_t ws_size,
                              hipStream_t stream)
{
    (void)in_sizes; (void)n_in; (void)out_size; (void)ws_size;

    const int*   sentences = (const int*)  d_in[0];
    const int*   tags      = (const int*)  d_in[1];
    const float* embedding = (const float*)d_in[3];
    const float* W_ih_f    = (const float*)d_in[4];
    const float* W_hh_f    = (const float*)d_in[5];
    const float* b_ih_f    = (const float*)d_in[6];
    const float* b_hh_f    = (const float*)d_in[7];
    const float* W_ih_b    = (const float*)d_in[8];
    const float* W_hh_b    = (const float*)d_in[9];
    const float* b_ih_b    = (const float*)d_in[10];
    const float* b_hh_b    = (const float*)d_in[11];
    const float* W_out     = (const float*)d_in[12];
    const float* b_out     = (const float*)d_in[13];
    const float* trans     = (const float*)d_in[14];
    const float* h0        = (const float*)d_in[15];
    const float* c0        = (const float*)d_in[16];
    float* out = (float*)d_out;

    // workspace carve-up (all chunks 256B aligned by construction)
    char* ws = (char*)d_ws;
    size_t off = 0;
    unsigned short* WihX = (unsigned short*)(ws + off); off += (size_t)2 * G4H * HH * 2;  // 1 MB
    unsigned short* WhhP = (unsigned short*)(ws + off); off += (size_t)2 * G4H * HH * 2;  // 1 MB
    unsigned short* Xb   = (unsigned short*)(ws + off); off += (size_t)BB * TT * EE * 2;  // 4 MB
    float* XGT   = (float*)(ws + off); off += (size_t)2 * TT * G4H * BB * 4;              // 67 MB
    float* HS    = (float*)(ws + off); off += (size_t)2 * TT * HH * BB * 4;               // 16.8 MB
    float* feats = (float*)(ws + off); off += (size_t)BB * TT * NT * 4;                   // 393 KB
    float* partial = (float*)(ws + off); off += 256;

    // K0: conversions + W_hh fragment repack + gather (3145728 work items)
    k0_prep<<<12288, 256, 0, stream>>>(sentences, embedding,
                                       W_ih_f, W_hh_f, W_ih_b, W_hh_b,
                                       WihX, WhhP, Xb);

    // K1: big input-projection GEMM (65536 wave-tiles, 8 waves/block)
    k1_input_gemm<<<8192, 256, 0, stream>>>(Xb, WihX,
                                            b_ih_f, b_hh_f, b_ih_b, b_hh_b, XGT);

    // K2: sequential recurrence, one block per direction
    k2_lstm_rec<<<2, 1024, 0, stream>>>(WhhP, XGT, h0, c0, HS);

    // K3: tag projection
    k3_feats<<<TT * NT, 32, 0, stream>>>(HS, W_out, b_out, feats);

    // K4: per-batch CRF
    k4_crf<<<BB, 32, 0, stream>>>(feats, trans, tags, partial);

    // K5: final reduction to scalar
    k5_reduce<<<1, 32, 0, stream>>>(partial, out);
}
